// Attention_60155311948227
// MI455X (gfx1250) — matrix operand order repaired and checked by reading, not compiled, not run on hardware
//
#include <hip/hip_runtime.h>

// ---------------------------------------------------------------------------
// MI455X (gfx1250) implementation. All GEMMs run on v_wmma_f32_16x16x32_bf16
// (wave32, 16x16 tiles, f32 accumulate). fp32 inputs are converted once to
// bf16; BN / softmax / L2-norm scalars stay fp32. L2 normalization is folded
// into the attention logits (1/|q| 1/|k|) and softmax weights (1/|v|), so the
// Q/K/V FC kernels store pre-norm bf16 activations + per-row inverse norms.
// A-row-block staging uses GLOBAL_LOAD_ASYNC_TO_LDS when the toolchain
// exposes the builtin (ASYNCcnt-tracked, no VGPR round-trip), else the plain
// global->VGPR->ds path.
// ---------------------------------------------------------------------------

typedef unsigned int   u32;
typedef unsigned short u16;
typedef __attribute__((ext_vector_type(16))) __bf16 v16bf;
typedef __attribute__((ext_vector_type(8)))  float  v8f;
typedef __attribute__((ext_vector_type(4)))  u32    v4u;
typedef __attribute__((ext_vector_type(8)))  u32    v8u;
typedef int v4i_vs __attribute__((vector_size(16)));   // matches builtin's pointee type

#define BLOCK  256
#define NWAVE  8
#define KCHUNK 1024   // k-columns of the A row-block staged in LDS (16*1024*2B = 32KB)

#if defined(__has_builtin)
#  if __has_builtin(__builtin_amdgcn_global_load_async_to_lds_b128) && \
      __has_builtin(__builtin_amdgcn_s_wait_asynccnt)
#    define USE_ASYNC_LDS 1
#  endif
#endif

__device__ __forceinline__ u16 f2bf(float f) {          // round-to-nearest-even
  u32 u = __builtin_bit_cast(u32, f);
  return (u16)((u + 0x7FFFu + ((u >> 16) & 1u)) >> 16);
}

__device__ __forceinline__ v16bf pack16(v4u lo, v4u hi) {
  v8u t;
  t[0] = lo[0]; t[1] = lo[1]; t[2] = lo[2]; t[3] = lo[3];
  t[4] = hi[0]; t[5] = hi[1]; t[6] = hi[2]; t[7] = hi[3];
  return __builtin_bit_cast(v16bf, t);
}

// Stage one 16B chunk global->LDS (async path when available).
__device__ __forceinline__ void stage16(const u16* g, u16* l) {
#if defined(USE_ASYNC_LDS)
  __builtin_amdgcn_global_load_async_to_lds_b128(
      (__attribute__((address_space(1))) v4i_vs*)(u16*)g,
      (__attribute__((address_space(3))) v4i_vs*)l, 0, 0);
#else
  *(v4u*)l = *(const v4u*)g;
#endif
}

__device__ __forceinline__ void stage_fence() {
#if defined(USE_ASYNC_LDS)
  __builtin_amdgcn_s_wait_asynccnt(0);
#endif
}

// ------------------------------ elementwise --------------------------------

__global__ void cvt_kernel(const float* __restrict__ in, u16* __restrict__ out, int n) {
  int i = blockIdx.x * BLOCK + threadIdx.x;
  if (i < n) out[i] = f2bf(in[i]);
}

// mean over trailing 7x7 window, emit bf16 A_pool
__global__ void pool_kernel(const float* __restrict__ x, u16* __restrict__ out, int n) {
  int i = blockIdx.x * BLOCK + threadIdx.x;
  if (i >= n) return;
  const float* p = x + (long long)i * 49;
  float s = 0.f;
  for (int j = 0; j < 49; ++j) s += p[j];
  out[i] = f2bf(s * (1.f / 49.f));
}

// out[i] = x[i] + f[i/49]   (broadcast f_context over the 7x7 window)
__global__ void bcast_add_kernel(const float* __restrict__ x, const float* __restrict__ f,
                                 float* __restrict__ out, long long n) {
  long long i = (long long)blockIdx.x * BLOCK + threadIdx.x;
  if (i < n) out[i] = x[i] + f[i / 49];
}

// ------------------------------ FC / GEMM ----------------------------------
// out[M,N] = epilogue( A[M,Kd] @ W[N,Kd]^T ).  Row block of 16 per workgroup,
// 8 waves each computing a 16x16 WMMA tile per N-pass (128 cols / pass).
// A row block staged in LDS in 32KB chunks; B streamed from global (L2-reuse).
// WMMA bf16 operand layouts per CDNA5 ISA 7.12.2:
//   A: lane l (row = l&15): VGPR0-3 = K (l>>4)*8 .. +7, VGPR4-7 = K 16+(l>>4)*8 ..
//   B: lane l (col = l&15): VGPR0-3 = K (l>>4)*8 .. +7, VGPR4-7 = K 16+(l>>4)*8 .. (same as A)
//   C: lane l: VGPR i -> row i + 8*(l>>4), col l&15
template<bool BNRELU, bool NORM, bool TRANS, bool F32OUT>
__global__ void __launch_bounds__(BLOCK)
fc_kernel(const u16* __restrict__ A, const u16* __restrict__ W,
          const float* __restrict__ bias, const float* __restrict__ gam,
          const float* __restrict__ bet,  const float* __restrict__ mea,
          const float* __restrict__ var,
          void* __restrict__ outp, float* __restrict__ invnorm,
          int M, int N, int Kd, int rows_per_batch, long long w_bstride)
{
  __shared__ u16   As[16 * KCHUNK];
  __shared__ float ssq[16];
  (void)M;

  const int tid  = threadIdx.x;
  const int wave = tid >> 5;
  const int lane = tid & 31;
  const int hf   = lane >> 4;
  const int l16  = lane & 15;
  const int r0   = blockIdx.x * 16;
  const int batch = r0 / rows_per_batch;
  const u16* Wb = W + (long long)batch * w_bstride;
  const int nchunk = Kd / KCHUNK;

  if (NORM && tid < 16) ssq[tid] = 0.f;

  for (int n0 = 0; n0 < N; n0 += NWAVE * 16) {
    const int nw = n0 + wave * 16;
    v8f acc = {};
    for (int kc = 0; kc < nchunk; ++kc) {
      if (nchunk > 1 || n0 == 0) {      // single-chunk case: stage once
        __syncthreads();
        for (int i = tid; i < 16 * (KCHUNK / 8); i += BLOCK) {
          int row = i >> 7, c8 = i & 127;
          stage16(A + (long long)(r0 + row) * Kd + kc * KCHUNK + c8 * 8,
                  &As[row * KCHUNK + c8 * 8]);
        }
        stage_fence();
        __syncthreads();
      }
      const long long brow = (long long)(nw + l16) * Kd + kc * KCHUNK;
      #pragma unroll 8   // 8-deep WMMA/load pipeline, keep VGPRs under 256
      for (int ks = 0; ks < KCHUNK; ks += 32) {
        v4u a0 = *(const v4u*)&As[l16 * KCHUNK + ks + hf * 8];
        v4u a1 = *(const v4u*)&As[l16 * KCHUNK + ks + 16 + hf * 8];
        const u16* bp = Wb + brow + ks + hf * 8;
        v4u b0 = *(const v4u*)bp;
        v4u b1 = *(const v4u*)(bp + 16);
        acc = __builtin_amdgcn_wmma_f32_16x16x32_bf16(false, pack16(a0, a1),
                                                      false, pack16(b0, b1),
                                                      (short)0, acc, false, false);
      }
    }
    // ---- epilogue: bias -> ReLU -> BN (eval) ----
    const int col = nw + l16;
    float bi = 0.f, scale = 1.f, shift = 0.f;
    if (BNRELU) {
      bi = bias[col];
      float s = gam[col] * rsqrtf(var[col] + 1e-5f);
      scale = s;
      shift = bet[col] - mea[col] * s;
    }
    float vals[8];
    #pragma unroll
    for (int i = 0; i < 8; ++i) {
      float y = acc[i];
      if (BNRELU) y = fmaxf(y + bi, 0.f) * scale + shift;
      vals[i] = y;
    }
    if (NORM) {   // per-row sum of squares: butterfly over the 16-lane group, LDS atomic
      #pragma unroll
      for (int i = 0; i < 8; ++i) {
        float s = vals[i] * vals[i];
        #pragma unroll
        for (int o = 8; o >= 1; o >>= 1) s += __shfl_xor(s, o, 16);
        if (l16 == 0) atomicAdd(&ssq[i + 8 * hf], s);
      }
    }
    if (F32OUT) {
      float* o = (float*)outp;
      #pragma unroll
      for (int i = 0; i < 8; ++i)
        o[(long long)(r0 + i + 8 * hf) * N + col] = vals[i];
    } else if (TRANS) {   // store out[b][col][m]: 8 consecutive m -> one b128 store
      v4u pk;
      #pragma unroll
      for (int i = 0; i < 4; ++i)
        pk[i] = (u32)f2bf(vals[2 * i]) | ((u32)f2bf(vals[2 * i + 1]) << 16);
      u16* o = (u16*)outp + (long long)batch * N * rows_per_batch
             + (long long)col * rows_per_batch + (r0 % rows_per_batch) + 8 * hf;
      *(v4u*)o = pk;
    } else {
      u16* o = (u16*)outp;
      #pragma unroll
      for (int i = 0; i < 8; ++i)
        o[(long long)(r0 + i + 8 * hf) * N + col] = f2bf(vals[i]);
    }
  }
  if (NORM) {
    __syncthreads();
    if (tid < 16) {
      float nrm = sqrtf(ssq[tid]);
      invnorm[r0 + tid] = 1.f / fmaxf(nrm, 1e-12f);   // matches y/max(|y|,1e-12)
    }
  }
}

// ------------------------------ attention ----------------------------------
// One workgroup = 16 q-rows of one batch. Full 16x2048 logit panel kept in
// registers (16 v8f accumulators per wave, wave-interleaved m-tiles), then
// scale by invq*invk/TEMP, mask, two-level (shfl + LDS) max/sum reductions,
// emit softmax * invv as bf16 attention weights.
#define D1 2048
#define MM 2048
#define NT 16

__global__ void __launch_bounds__(BLOCK)
attn_kernel(const u16* __restrict__ Q, const u16* __restrict__ Kp,
            const float* __restrict__ invq, const float* __restrict__ invk,
            const float* __restrict__ invv, const int* __restrict__ nvalid,
            u16* __restrict__ Wout)
{
  __shared__ u16   Qs[16 * KCHUNK];
  __shared__ float red[NWAVE][16];

  const int tid  = threadIdx.x;
  const int wave = tid >> 5, lane = tid & 31;
  const int hf   = lane >> 4, l16 = lane & 15;
  const int b    = blockIdx.x >> 2;
  const int n0   = (blockIdx.x & 3) * 16;
  const int nv   = nvalid[b];

  v8f vzero = {};
  v8f accs[NT];
  #pragma unroll
  for (int t = 0; t < NT; ++t) accs[t] = vzero;

  for (int kc = 0; kc < D1 / KCHUNK; ++kc) {
    __syncthreads();
    for (int i = tid; i < 16 * (KCHUNK / 8); i += BLOCK) {
      int row = i >> 7, c8 = i & 127;
      stage16(Q + (long long)(b * 64 + n0 + row) * D1 + kc * KCHUNK + c8 * 8,
              &Qs[row * KCHUNK + c8 * 8]);
    }
    stage_fence();
    __syncthreads();
    #pragma unroll
    for (int t = 0; t < NT; ++t) {
      const int mt = (t * NWAVE + wave) * 16;
      const long long brow = (long long)(b * MM + mt + l16) * D1 + kc * KCHUNK;
      #pragma unroll 8
      for (int ks = 0; ks < KCHUNK; ks += 32) {
        v4u a0 = *(const v4u*)&Qs[l16 * KCHUNK + ks + hf * 8];
        v4u a1 = *(const v4u*)&Qs[l16 * KCHUNK + ks + 16 + hf * 8];
        const u16* bp = Kp + brow + ks + hf * 8;
        v4u b0 = *(const v4u*)bp;
        v4u b1 = *(const v4u*)(bp + 16);
        accs[t] = __builtin_amdgcn_wmma_f32_16x16x32_bf16(false, pack16(a0, a1),
                                                          false, pack16(b0, b1),
                                                          (short)0, accs[t], false, false);
      }
    }
  }

  const float NEG = -__builtin_inff();
  float iq[8];
  #pragma unroll
  for (int i = 0; i < 8; ++i) iq[i] = invq[b * 64 + n0 + i + 8 * hf] * 100.f;  // 1/TEMP

  float rmax[8];
  #pragma unroll
  for (int i = 0; i < 8; ++i) rmax[i] = NEG;
  #pragma unroll
  for (int t = 0; t < NT; ++t) {
    const int mcol = (t * NWAVE + wave) * 16 + l16;
    const float kk = invk[b * MM + mcol];
    const bool ok = mcol < nv;
    #pragma unroll
    for (int i = 0; i < 8; ++i) {
      float v = ok ? accs[t][i] * iq[i] * kk : NEG;
      accs[t][i] = v;
      rmax[i] = fmaxf(rmax[i], v);
    }
  }
  #pragma unroll
  for (int i = 0; i < 8; ++i) {
    #pragma unroll
    for (int o = 8; o >= 1; o >>= 1) rmax[i] = fmaxf(rmax[i], __shfl_xor(rmax[i], o, 16));
  }
  __syncthreads();
  if (l16 == 0) {
    #pragma unroll
    for (int i = 0; i < 8; ++i) red[wave][i + 8 * hf] = rmax[i];
  }
  __syncthreads();
  #pragma unroll
  for (int i = 0; i < 8; ++i) {
    float mx = NEG;
    #pragma unroll
    for (int w = 0; w < NWAVE; ++w) mx = fmaxf(mx, red[w][i + 8 * hf]);
    rmax[i] = mx;
  }

  float rsum[8];
  #pragma unroll
  for (int i = 0; i < 8; ++i) rsum[i] = 0.f;
  #pragma unroll
  for (int t = 0; t < NT; ++t) {
    #pragma unroll
    for (int i = 0; i < 8; ++i) {
      float e = __expf(accs[t][i] - rmax[i]);   // exp(-inf)=0 handles the mask
      accs[t][i] = e;
      rsum[i] += e;
    }
  }
  #pragma unroll
  for (int i = 0; i < 8; ++i) {
    #pragma unroll
    for (int o = 8; o >= 1; o >>= 1) rsum[i] += __shfl_xor(rsum[i], o, 16);
  }
  __syncthreads();
  if (l16 == 0) {
    #pragma unroll
    for (int i = 0; i < 8; ++i) red[wave][i + 8 * hf] = rsum[i];
  }
  __syncthreads();
  #pragma unroll
  for (int i = 0; i < 8; ++i) {
    float s = 0.f;
    #pragma unroll
    for (int w = 0; w < NWAVE; ++w) s += red[w][i + 8 * hf];
    rsum[i] = 1.f / s;
  }

  #pragma unroll
  for (int t = 0; t < NT; ++t) {
    const int mcol = (t * NWAVE + wave) * 16 + l16;
    const float vv = invv[b * MM + mcol];       // folds V's L2 normalization
    #pragma unroll
    for (int i = 0; i < 8; ++i)
      Wout[(long long)(b * 64 + n0 + i + 8 * hf) * MM + mcol] = f2bf(accs[t][i] * rsum[i] * vv);
  }
}

// ------------------------------ launcher -----------------------------------

extern "C" void kernel_launch(void* const* d_in, const int* in_sizes, int n_in,
                              void* d_out, int out_size, void* d_ws, size_t ws_size,
                              hipStream_t stream)
{
  (void)in_sizes; (void)n_in; (void)out_size; (void)ws_size;

  const float* x   = (const float*)d_in[0];
  const float* xc  = (const float*)d_in[1];
  const int*   nv  = (const int*)d_in[2];
  const float* qW  = (const float*)d_in[3];
  const float* qb  = (const float*)d_in[4];
  const float* qg  = (const float*)d_in[5];
  const float* qbe = (const float*)d_in[6];
  const float* qm  = (const float*)d_in[7];
  const float* qv  = (const float*)d_in[8];
  const float* kW  = (const float*)d_in[9];
  const float* kb  = (const float*)d_in[10];
  const float* kg  = (const float*)d_in[11];
  const float* kbe = (const float*)d_in[12];
  const float* km  = (const float*)d_in[13];
  const float* kvv = (const float*)d_in[14];
  const float* vW  = (const float*)d_in[15];
  const float* vb  = (const float*)d_in[16];
  const float* vg  = (const float*)d_in[17];
  const float* vbe = (const float*)d_in[18];
  const float* vm  = (const float*)d_in[19];
  const float* vva = (const float*)d_in[20];
  const float* fW  = (const float*)d_in[21];
  const float* fb  = (const float*)d_in[22];
  const float* fg  = (const float*)d_in[23];
  const float* fbe = (const float*)d_in[24];
  const float* fm  = (const float*)d_in[25];
  const float* fva = (const float*)d_in[26];
  float* out = (float*)d_out;

  char* wp = (char*)d_ws;
  auto alloc = [&](size_t bytes) -> void* {
    void* p = (void*)wp;
    wp += (bytes + 255) & ~(size_t)255;
    return p;
  };
  u16*   qWb  = (u16*)alloc(2048ull * 1024 * 2);
  u16*   kWb  = (u16*)alloc(2048ull * 2048 * 2);
  u16*   vWb  = (u16*)alloc(2048ull * 2048 * 2);
  u16*   fWb  = (u16*)alloc(1024ull * 2048 * 2);
  u16*   xcb  = (u16*)alloc(8ull * 2048 * 2048 * 2);
  u16*   Apb  = (u16*)alloc(512ull * 1024 * 2);
  u16*   qpre = (u16*)alloc(512ull * 2048 * 2);
  float* invq = (float*)alloc(512 * 4);
  u16*   keyb = (u16*)alloc(16384ull * 2048 * 2);
  float* invk = (float*)alloc(16384 * 4);
  u16*   valT = (u16*)alloc(16384ull * 2048 * 2);   // [b][d2][m]
  float* invv = (float*)alloc(16384 * 4);
  u16*   wbuf = (u16*)alloc(512ull * 2048 * 2);     // softmax weights * invv
  u16*   wvb  = (u16*)alloc(512ull * 2048 * 2);
  float* fctx = (float*)alloc(512ull * 1024 * 4);

  // bf16 conversions + pooling
  cvt_kernel<<<2048 * 1024 / BLOCK, BLOCK, 0, stream>>>(qW, qWb, 2048 * 1024);
  cvt_kernel<<<2048 * 2048 / BLOCK, BLOCK, 0, stream>>>(kW, kWb, 2048 * 2048);
  cvt_kernel<<<2048 * 2048 / BLOCK, BLOCK, 0, stream>>>(vW, vWb, 2048 * 2048);
  cvt_kernel<<<1024 * 2048 / BLOCK, BLOCK, 0, stream>>>(fW, fWb, 1024 * 2048);
  cvt_kernel<<<33554432 / BLOCK, BLOCK, 0, stream>>>(xc, xcb, 33554432);
  pool_kernel<<<524288 / BLOCK, BLOCK, 0, stream>>>(x, Apb, 524288);

  // Q / K / V projections (ReLU+BN, pre-norm bf16 + inverse row norms)
  fc_kernel<true, true, false, false><<<32, BLOCK, 0, stream>>>(
      Apb, qWb, qb, qg, qbe, qm, qv, qpre, invq, 512, 2048, 1024, 512, 0);
  fc_kernel<true, true, false, false><<<1024, BLOCK, 0, stream>>>(
      xcb, kWb, kb, kg, kbe, km, kvv, keyb, invk, 16384, 2048, 2048, 2048, 0);
  fc_kernel<true, true, true, false><<<1024, BLOCK, 0, stream>>>(
      xcb, vWb, vb, vg, vbe, vm, vva, valT, invv, 16384, 2048, 2048, 2048, 0);

  // QK^T + mask + softmax (norms and 1/TEMP folded), weights scaled by invv
  attn_kernel<<<32, BLOCK, 0, stream>>>(qpre, keyb, invq, invk, invv, nv, wbuf);

  // WV: A = weights [b*64, m], W = valT[b] [d2, m] (per-batch W stride)
  fc_kernel<false, false, false, false><<<32, BLOCK, 0, stream>>>(
      wbuf, valT, nullptr, nullptr, nullptr, nullptr, nullptr, wvb, nullptr,
      512, 2048, 2048, 64, 2048ll * 2048);

  // final FC (ReLU+BN, f32 out), then broadcast-add over the 7x7 window
  fc_kernel<true, false, false, true><<<32, BLOCK, 0, stream>>>(
      wvb, fWb, fb, fg, fbe, fm, fva, fctx, nullptr, 512, 1024, 2048, 512, 0);
  bcast_add_kernel<<<25690112 / BLOCK, BLOCK, 0, stream>>>(x, fctx, out, 25690112ll);
}
